// LSTMAttentionCell_61237643706893
// MI455X (gfx1250) — matrix operand order repaired and checked by reading, not compiled, not run on hardware
//
#include <hip/hip_runtime.h>

// ---------------------------------------------------------------------------
// Graves attention-LSTM scan for MI455X (gfx1250, wave32, WMMA).
//
//   grid = 8 batch-tiles (16 rows) x 15 column-groups (32 cells) = 120 WGs
//   of 256 threads (8 waves), ~1 per WGP.
//   - Each WG owns 128 gate columns (32 cells x {i,f,g,o}) of z = A @ W.
//   - A = [x_t(3) | w(80) | pad(13) | h(480)] as 16x576 bf16 in LDS.
//   - Main-weight B-fragments resident in VGPRs (18 x v16bf = 144 VGPRs);
//     window weights in LDS (30KB) to keep pressure < 256 VGPRs.
//   - GEMMs are software-pipelined via __builtin_amdgcn_sched_group_barrier:
//     A-fragment DS reads run 3-4 tiles ahead of the consuming WMMA so the
//     inserted waits are partial (dscnt<=6) instead of 0.
//   - c private per WG; h exchanged per step via ping-pong L2 scratch +
//     per-group monotonic atomic barrier (15 co-resident WGs per batch tile).
//   - Window GEMM + mixture + w = phi @ sentence recomputed redundantly per
//     WG; sentence cached in LDS as bf16 (160KB; CDNA5 allows 320KB/WG).
// ---------------------------------------------------------------------------

typedef __attribute__((ext_vector_type(16))) __bf16 v16bf;
typedef __attribute__((ext_vector_type(8)))  __bf16 v8bf;
typedef __attribute__((ext_vector_type(8)))  float  v8f;

#define NUM_CHARS 80
#define KMIX      10
#define HDIM      480      // num_cells + num_chars
#define ZDIM      1920     // 4*HDIM
#define TSTEPS    800
#define ULEN      64
#define INDIM     3
#define DIN       83       // INDIM + NUM_CHARS
#define FOUT      570      // HDIM + KMIX + NUM_CHARS

#define CG   15            // column groups per batch tile (480/32)
#define BT   8             // batch tiles (128/16)
#define NT   8             // 16-wide n-tiles per group (128 cols)
#define KT   18            // k-tiles of 32 (576 = 96 + 480)
#define KTH  15            // k-tiles covering the h part
#define KX   96            // K offset of h inside A
#define APITCH 584         // A pitch (bf16): 292 dwords/row == 36 mod 64 banks
#define ZPITCH 132         // z pitch (f32)
#define WPITCH 36          // win pitch (f32)

// sched_group_barrier masks
#define SGB_MFMA   0x008   // MFMA/WMMA
#define SGB_DSREAD 0x100   // DS read

// LDS carve-up (bytes)
#define SM_SENT 0                         // 16*64*80 bf16 = 163840
#define SM_A    163840                    // 16*584 bf16   = 18688
#define SM_Z    (SM_A + 18688)            // 16*132 f32    = 8448
#define SM_C    (SM_Z + 8448)             // 16*32 f32     = 2048
#define SM_WIN  (SM_C + 2048)             // 16*36 f32     = 2304
#define SM_BETA (SM_WIN + 2304)           // 16*10 f32     = 640
#define SM_KAP  (SM_BETA + 640)           // 16*10 f32     = 640
#define SM_PHI  (SM_KAP + 640)            // 16*64 f32     = 4096
#define SM_SLEN (SM_PHI + 4096)           // 16 i32        = 64
#define SM_BWINL (SM_SLEN + 64)           // 15*2*512 bf16 = 30720 (32B aligned)
#define SMEM_BYTES (SM_BWINL + 30720)     // 231488

// d_ws layout (bytes)
#define WS_BARS  0                                   // 8 counters @ 64B stride
#define WS_HSCR  512                                 // 8*2*16*480 bf16 = 245760
#define WS_BMAIN (WS_HSCR + 245760)                  // 15*8*18*512 bf16 = 2211840
#define WS_BWIN  (WS_BMAIN + 2211840)                // 15*2*512 bf16 = 30720

__device__ __forceinline__ float sigmoid_f(float z) {
  return 1.0f / (1.0f + __expf(-z));
}
__device__ __forceinline__ float tanh_f(float z) {
  float e = __expf(2.0f * z);          // robust at +-inf
  return 1.0f - 2.0f / (e + 1.0f);
}

// --------------------------------------------------------------------------
// Prep: pack W_k (83x1920) + W_r (480x1920) + W_win (480x30) into bf16
// B-fragments [g][nt][kt][lane][16e] in this file's contiguous k order (same as load_afrag)
// (lanes 0-15: K=0..15, lanes 16-31: K=16..31, N = lane%16); zero barriers.
// --------------------------------------------------------------------------
__global__ __launch_bounds__(256)
void prep_pack(const float* __restrict__ Wk, const float* __restrict__ Wr,
               const float* __restrict__ Wwin,
               __bf16* __restrict__ Bmain, __bf16* __restrict__ Bwin,
               unsigned* __restrict__ bars)
{
  int tid = blockIdx.x * 256 + threadIdx.x;
  if (tid < BT) bars[tid * 16] = 0u;

  const int MAIN = CG * NT * KT * 512;
  if (tid < MAIN) {
    int idx = tid;
    int e    = idx & 15; idx >>= 4;
    int lane = idx & 31; idx >>= 5;
    int kt   = idx % KT; idx /= KT;
    int nt   = idx % NT; idx /= NT;
    int g    = idx;
    int K    = kt * 32 + ((lane >> 4) << 4) + e;
    int col  = (nt >> 1) * HDIM + 32 * g + ((nt & 1) << 4) + (lane & 15);
    float v = 0.0f;
    if (K < DIN)      v = Wk[K * ZDIM + col];
    else if (K >= KX) v = Wr[(K - KX) * ZDIM + col];
    Bmain[tid] = (__bf16)v;
  } else if (tid < MAIN + KTH * 2 * 512) {
    int idx = tid - MAIN;
    int e    = idx & 15; idx >>= 4;
    int lane = idx & 31; idx >>= 5;
    int nt   = idx & 1;  idx >>= 1;
    int kt   = idx;
    int K    = kt * 32 + ((lane >> 4) << 4) + e;   // 0..479
    int col  = (nt << 4) + (lane & 15);
    float v  = (col < 3 * KMIX) ? Wwin[K * (3 * KMIX) + col] : 0.0f;
    Bwin[tid - MAIN] = (__bf16)v;
  }
}

__device__ __forceinline__ v16bf load_afrag(const __bf16* Alds, int lane, int kbase) {
  // 16-bit A fragment: lane L -> row L%16; K = 0..15 (L<16) or
  // 16..31 (L>=16) relative to kbase.  Two 16B LDS reads.
  const __bf16* ar = Alds + (lane & 15) * APITCH + kbase + ((lane >> 4) << 4);
  v8bf lo = *(const v8bf*)ar;
  v8bf hi = *(const v8bf*)(ar + 8);
  return __builtin_shufflevector(lo, hi, 0,1,2,3,4,5,6,7,8,9,10,11,12,13,14,15);
}

// --------------------------------------------------------------------------
// Main scan kernel: 120 WGs x 256 threads, 800 sequential steps.
// --------------------------------------------------------------------------
__global__ __launch_bounds__(256)
void graves_attn_lstm(const float* __restrict__ x,
                      const float* __restrict__ sentence,
                      const int*   __restrict__ slen_g,
                      const float* __restrict__ b_lstm,
                      const float* __restrict__ b_win,
                      const __bf16* __restrict__ Bmain,
                      const __bf16* __restrict__ Bwin,
                      __bf16* __restrict__ hscr,
                      unsigned* __restrict__ bars,
                      float* __restrict__ out)
{
  extern __shared__ char smem[];
  __bf16* sentL = (__bf16*)(smem + SM_SENT);
  __bf16* A     = (__bf16*)(smem + SM_A);
  float*  zbuf  = (float*)(smem + SM_Z);
  float*  cbuf  = (float*)(smem + SM_C);
  float*  winL  = (float*)(smem + SM_WIN);
  float*  betaL = (float*)(smem + SM_BETA);
  float*  kapL  = (float*)(smem + SM_KAP);
  float*  phiL  = (float*)(smem + SM_PHI);
  int*    slenL = (int*)  (smem + SM_SLEN);
  __bf16* bwinL = (__bf16*)(smem + SM_BWINL);

  const int tid  = threadIdx.x;
  const int wave = tid >> 5;
  const int lane = tid & 31;
  const int g    = blockIdx.x % CG;   // column group (32 cells)
  const int bt   = blockIdx.x / CG;   // batch tile (16 rows)

  // ---- one-time init -----------------------------------------------------
  for (int i = tid; i < 16 * APITCH; i += 256) A[i] = (__bf16)0.0f;   // w=h=0
  for (int i = tid; i < 512; i += 256) cbuf[i] = 0.0f;                 // c=0
  for (int i = tid; i < 160; i += 256) kapL[i] = 0.0f;                 // kappa=0
  if (tid < 16) slenL[tid] = slen_g[bt * 16 + tid];
  for (int i = tid; i < 16 * ULEN * NUM_CHARS; i += 256)
    sentL[i] = (__bf16)sentence[(size_t)bt * 16 * ULEN * NUM_CHARS + i];
  // window weight fragments -> LDS (1920 x uint4)
  {
    const uint4* src = (const uint4*)Bwin;
    uint4* dst = (uint4*)bwinL;
    for (int i = tid; i < (KTH * 2 * 512) / 8; i += 256) dst[i] = src[i];
  }
  if (tid < 48) {                                                      // x_0
    int b = tid / 3, k = tid % 3;
    A[b * APITCH + k] = (__bf16)x[((size_t)(bt * 16 + b) * TSTEPS) * INDIM + k];
  }

  // per-lane LSTM bias for this wave's n-tile (constant over time)
  const int zcol = (wave >> 1) * HDIM + 32 * g + ((wave & 1) << 4) + (lane & 15);
  const float bias = b_lstm[zcol];
  float bwin_lane = 0.0f;
  {
    int wcol = ((wave & 1) << 4) + (lane & 15);
    if (wcol < 3 * KMIX) bwin_lane = b_win[wcol];
  }

  // resident main-weight fragments: loaded from L2 once, live in VGPRs
  const __bf16* Bp = Bmain + ((size_t)(g * NT + wave) * KT) * 512;
  v16bf Breg[KT];
#pragma unroll
  for (int kt = 0; kt < KT; ++kt)
    Breg[kt] = *(const v16bf*)(Bp + (size_t)kt * 512 + lane * 16);

  __syncthreads();

  unsigned* mybar = bars + bt * 16;   // 64B-strided per-group counter

  for (int t = 0; t < TSTEPS; ++t) {
    // 1) z slice: 8 waves x 1 n-tile x 18 k-tiles.  Pipeline shape imposed
    //    with sched_group_barrier: 4 tiles of A-frag DS reads up front, then
    //    loads stay 4 tiles ahead of the consuming WMMA.
    v8f acc = {};
    {
#pragma unroll
      for (int kt = 0; kt < KT; ++kt) {
        v16bf af = load_afrag(A, lane, kt * 32);
        acc = __builtin_amdgcn_wmma_f32_16x16x32_bf16(
            false, af, false, Breg[kt], (short)0, acc, false, false);
      }
      // [DS x8] ([WMMA x1][DS x2]) x14 [WMMA x4]
      __builtin_amdgcn_sched_group_barrier(SGB_DSREAD, 8, 0);
#pragma unroll
      for (int i = 0; i < KT - 4; ++i) {
        __builtin_amdgcn_sched_group_barrier(SGB_MFMA, 1, 0);
        __builtin_amdgcn_sched_group_barrier(SGB_DSREAD, 2, 0);
      }
      __builtin_amdgcn_sched_group_barrier(SGB_MFMA, 4, 0);
    }
    {
      int nb = wave * 16 + (lane & 15), mo = (lane >> 4) << 3;
#pragma unroll
      for (int r = 0; r < 8; ++r)
        zbuf[(mo + r) * ZPITCH + nb] = acc[r] + bias;  // C/D: M=r(+8), N=lane%16
    }
    __syncthreads();

    // 2) gates -> c,h for this WG's 32 cells; publish h slice
    for (int i = tid; i < 512; i += 256) {
      int b = i >> 5, n = i & 31;
      float zi = zbuf[b * ZPITCH + n];
      float zf = zbuf[b * ZPITCH + 32 + n];
      float zg = zbuf[b * ZPITCH + 64 + n];
      float zo = zbuf[b * ZPITCH + 96 + n];
      float cn = sigmoid_f(zf) * cbuf[i] + sigmoid_f(zi) * tanh_f(zg);
      cbuf[i] = cn;
      float hn = sigmoid_f(zo) * tanh_f(cn);
      int cell = 32 * g + n;
      out[((size_t)(bt * 16 + b) * TSTEPS + t) * FOUT + cell] = hn;
      hscr[((size_t)(bt * 2 + (t & 1)) * 16 + b) * HDIM + cell] = (__bf16)hn;
    }
    __syncthreads();

    // 3) stage x_{t+1} now so its load latency overlaps the barrier spin
    if (t + 1 < TSTEPS && tid < 48) {
      int b = tid / 3, k = tid % 3;
      A[b * APITCH + k] =
        (__bf16)x[((size_t)(bt * 16 + b) * TSTEPS + (t + 1)) * INDIM + k];
    }

    // 4) group barrier (15 WGs, monotonic counter; h scratch is ping-ponged
    //    so one barrier per step suffices)
    if (tid == 0) {
      __hip_atomic_fetch_add(mybar, 1u, __ATOMIC_RELEASE,
                             __HIP_MEMORY_SCOPE_AGENT);
      unsigned tgt = (unsigned)CG * (unsigned)(t + 1);
      while (__hip_atomic_load(mybar, __ATOMIC_ACQUIRE,
                               __HIP_MEMORY_SCOPE_AGENT) < tgt)
        __builtin_amdgcn_s_sleep(1);
    }
    __syncthreads();

    // 5) gather full h (16x480 bf16 = 960 x b128) into A[:,96:576]
    {
      const uint4* hs =
        (const uint4*)(hscr + (size_t)(bt * 2 + (t & 1)) * 16 * HDIM);
      for (int i = tid; i < 16 * (HDIM / 8); i += 256) {   // 960
        int b = i / (HDIM / 8), jc = i % (HDIM / 8);
        ((uint4*)((char*)(A + b * APITCH) + KX * 2))[jc] = hs[i];
      }
    }
    __syncthreads();

    // 6) window GEMM: win = h @ W_win (+b_win), waves 0..1; A and B frags
    //    both from LDS (4 DS reads/tile), pipelined 3 tiles ahead
    if (wave < 2) {
      v8f wacc = {};
      const __bf16* bw = bwinL + (size_t)((wave & 1) * 32 + lane) * 16;
#pragma unroll
      for (int kt = 0; kt < KTH; ++kt) {
        v16bf af = load_afrag(A, lane, KX + kt * 32);
        v16bf bf = *(const v16bf*)(bw + (size_t)kt * 2 * 512);
        wacc = __builtin_amdgcn_wmma_f32_16x16x32_bf16(
            false, af, false, bf, (short)0, wacc, false, false);
      }
      // [DS x12] ([WMMA x1][DS x4]) x12 [WMMA x3]
      __builtin_amdgcn_sched_group_barrier(SGB_DSREAD, 12, 0);
#pragma unroll
      for (int i = 0; i < KTH - 3; ++i) {
        __builtin_amdgcn_sched_group_barrier(SGB_MFMA, 1, 0);
        __builtin_amdgcn_sched_group_barrier(SGB_DSREAD, 4, 0);
      }
      __builtin_amdgcn_sched_group_barrier(SGB_MFMA, 3, 0);

      int nb = wave * 16 + (lane & 15), mo = (lane >> 4) << 3;
#pragma unroll
      for (int r = 0; r < 8; ++r)
        winL[(mo + r) * WPITCH + nb] = wacc[r] + bwin_lane;
    }
    __syncthreads();

    // 7) kappa += exp(k_hat); beta = exp(b_hat)
    if (tid < 160) {
      int b = tid / KMIX, k = tid % KMIX;
      betaL[tid] = __expf(winL[b * WPITCH + KMIX + k]);
      float kap = kapL[tid] + __expf(winL[b * WPITCH + 2 * KMIX + k]);
      kapL[tid] = kap;
      if (g == 0)
        out[((size_t)(bt * 16 + b) * TSTEPS + t) * FOUT + HDIM + k] = kap;
    }
    __syncthreads();

    // 8) phi[b,u] = sum_k exp(a_hat - beta*(kappa-u-1)^2), masked by length
    for (int i = tid; i < 16 * ULEN; i += 256) {
      int b = i >> 6, u = i & 63;
      float s = 0.0f;
      if (u < slenL[b]) {
#pragma unroll
        for (int k = 0; k < KMIX; ++k) {
          float d = kapL[b * KMIX + k] - (float)(u + 1);
          s += __expf(winL[b * WPITCH + k] - betaL[b * KMIX + k] * d * d);
        }
      }
      phiL[i] = s;
    }
    __syncthreads();

    // 9) w = phi @ sentence (per-row), stage into A[:,3:83] for next step
    for (int i = tid; i < 16 * NUM_CHARS; i += 256) {
      int b = i / NUM_CHARS, ch = i % NUM_CHARS;
      float s = 0.0f;
      for (int u = 0; u < ULEN; ++u)
        s += phiL[b * ULEN + u] * (float)sentL[(b * ULEN + u) * NUM_CHARS + ch];
      A[b * APITCH + INDIM + ch] = (__bf16)s;
      if (g == 0)
        out[((size_t)(bt * 16 + b) * TSTEPS + t) * FOUT + HDIM + KMIX + ch] = s;
    }
    __syncthreads();
  }
}

extern "C" void kernel_launch(void* const* d_in, const int* in_sizes, int n_in,
                              void* d_out, int out_size, void* d_ws, size_t ws_size,
                              hipStream_t stream) {
  (void)in_sizes; (void)n_in; (void)out_size; (void)ws_size;
  const float* x    = (const float*)d_in[0];
  const float* sent = (const float*)d_in[1];
  const int*   slen = (const int*)d_in[2];
  const float* Wk   = (const float*)d_in[3];
  const float* Wr   = (const float*)d_in[4];
  const float* bl   = (const float*)d_in[5];
  const float* Wwin = (const float*)d_in[6];
  const float* bwin = (const float*)d_in[7];
  float* out = (float*)d_out;

  char* ws = (char*)d_ws;
  unsigned* bars  = (unsigned*)(ws + WS_BARS);
  __bf16*   hscr  = (__bf16*)(ws + WS_HSCR);
  __bf16*   Bmain = (__bf16*)(ws + WS_BMAIN);
  __bf16*   Bwin  = (__bf16*)(ws + WS_BWIN);

  int prep_elems = CG * NT * KT * 512 + KTH * 2 * 512;
  prep_pack<<<(prep_elems + 255) / 256, 256, 0, stream>>>(
      Wk, Wr, Wwin, Bmain, Bwin, bars);

  (void)hipFuncSetAttribute((const void*)graves_attn_lstm,
                            hipFuncAttributeMaxDynamicSharedMemorySize,
                            (int)SMEM_BYTES);
  graves_attn_lstm<<<CG * BT, 256, SMEM_BYTES, stream>>>(
      x, sent, slen, bl, bwin, Bmain, Bwin, hscr, bars, out);
}
